// dummy_gnn_model_20203526160417
// MI455X (gfx1250) — matrix operand order repaired and checked by reading, not compiled, not run on hardware
//
#include <hip/hip_runtime.h>
#include <hip/hip_bf16.h>
#include <stdint.h>

typedef __attribute__((ext_vector_type(16))) __bf16  v16bf;
typedef __attribute__((ext_vector_type(8)))  float   v8f;
typedef __attribute__((ext_vector_type(8)))  unsigned int v8u;

#define N_NODES 50000
#define N_EDGES 640000

// ---------------- degree: one thread per edge ----------------
__global__ void deg_kernel(const int* __restrict__ dst, float* __restrict__ deg, int E) {
    int e = blockIdx.x * blockDim.x + threadIdx.x;
    if (e < E) atomicAdd(&deg[dst[e]], 1.0f);
}

// ---------------- f32 -> bf16 weight conversion ----------------
__global__ void cvt_bf16_kernel(const float* __restrict__ w, __hip_bfloat16* __restrict__ o, int n) {
    int i = blockIdx.x * blockDim.x + threadIdx.x;
    if (i < n) o[i] = __float2bfloat16(w[i]);
}

// ---------------- scatter: one wave32 per edge ----------------
// agg[dst] += h[src] * e_w  (coalesced loads, coalesced L2 atomics)
template <int DIN>
__global__ void scatter_kernel(const float* __restrict__ h, const float* __restrict__ ew,
                               const int* __restrict__ src, const int* __restrict__ dst,
                               float* __restrict__ agg, int E) {
    int wave = (blockIdx.x * blockDim.x + threadIdx.x) >> 5;
    int lane = threadIdx.x & 31;
    if (wave >= E) return;
    int s = src[wave];
    int d = dst[wave];
    float w = ew[wave];
    const float* hs = h + (size_t)s * DIN;
    float*       ad = agg + (size_t)d * DIN;
#pragma unroll
    for (int c = lane; c < DIN; c += 32)
        atomicAdd(&ad[c], hs[c] * w);
}

// ---------------- build X_bf16 = [agg * inv_deg | h] ----------------
template <int DIN>
__global__ void build_x_kernel(const float* __restrict__ agg, const float* __restrict__ h,
                               const float* __restrict__ deg, __hip_bfloat16* __restrict__ X, int N) {
    int tid = blockIdx.x * blockDim.x + threadIdx.x;
    const int KD = 2 * DIN;
    if (tid >= N * KD) return;
    int n = tid / KD;
    int c = tid - n * KD;
    float v;
    if (c < DIN) {
        float inv = 1.0f / fmaxf(deg[n], 1.0f);
        v = agg[(size_t)n * DIN + c] * inv;
    } else {
        v = h[(size_t)n * DIN + (c - DIN)];
    }
    X[tid] = __float2bfloat16(v);
}

// ---------------- WMMA GEMM: out[N, DOUT] = X[N, K] @ Wb[DOUT, K]^T + bias ----------------
// One wave computes a 16 x (16*NT) output slab via v_wmma_f32_16x16x32_bf16.
template <int K, int DOUT, int NT, bool RELU>
__global__ void gemm_wmma_kernel(const __hip_bfloat16* __restrict__ X,
                                 const __hip_bfloat16* __restrict__ Wb,
                                 const float* __restrict__ bias,
                                 float* __restrict__ out, int N) {
    int wid  = (blockIdx.x * blockDim.x + threadIdx.x) >> 5;
    int lane = threadIdx.x & 31;
    const int mtiles  = N >> 4;            // N divisible by 16 (50000 = 3125*16)
    const int ngroups = DOUT / (16 * NT);
    if (wid >= mtiles * ngroups) return;   // wave-uniform: EXEC stays all-1s for WMMA
    const int mt = wid / ngroups;
    const int ng = wid - mt * ngroups;
    const int mbase  = mt << 4;
    const int nbase0 = ng * 16 * NT;

    const int half = lane >> 4;            // K-half selector
    const int mrow = lane & 15;            // A-matrix row
    const int ncol = lane & 15;            // B/C/D column

    v8f acc[NT] = {};

    const uint32_t* Xu = reinterpret_cast<const uint32_t*>(X);   // 2 bf16 per dword
    const uint32_t* Wu = reinterpret_cast<const uint32_t*>(Wb);
    const size_t xrow = (size_t)(mbase + mrow) * (K / 2);

    for (int kk = 0; kk < K; kk += 32) {
        // A fragment: ISA 16-bit A 16x32 layout: VGPR v holds K = (v>>2)*16 + half*8 + (v&3)*2 (+1)
        v8u au;
#pragma unroll
        for (int v = 0; v < 8; v++) {
            int k = kk + ((v >> 2) << 4) + (half << 3) + ((v & 3) << 1);
            au[v] = Xu[xrow + (k >> 1)];
        }
        v16bf a = __builtin_bit_cast(v16bf, au);
#pragma unroll
        for (int t = 0; t < NT; t++) {
            const int nb = nbase0 + t * 16;
            // B fragment: same K order as A: VGPR v holds K = (v>>2)*16 + half*8 + (v&3)*2 (+1)
            const size_t wrow = (size_t)(nb + ncol) * (K / 2);
            v8u bu;
#pragma unroll
            for (int v = 0; v < 8; v++) {
                int k = kk + ((v >> 2) << 4) + (half << 3) + ((v & 3) << 1);
                bu[v] = Wu[wrow + (k >> 1)];
            }
            v16bf b = __builtin_bit_cast(v16bf, bu);
            acc[t] = __builtin_amdgcn_wmma_f32_16x16x32_bf16(
                false, a, false, b, (short)0, acc[t], false, false);
        }
    }

    // Epilogue: C/D f32 layout: VGPR r -> M = r + half*8, N = lane&15
#pragma unroll
    for (int t = 0; t < NT; t++) {
        const int nb = nbase0 + t * 16;
        const float bv = bias[nb + ncol];
#pragma unroll
        for (int r = 0; r < 8; r++) {
            int m = mbase + r + (half << 3);
            float v = acc[t][r] + bv;
            if (RELU) v = fmaxf(v, 0.0f);
            out[(size_t)m * DOUT + nb + ncol] = v;
        }
    }
}

static inline size_t align256(size_t x) { return (x + 255) & ~(size_t)255; }

extern "C" void kernel_launch(void* const* d_in, const int* in_sizes, int n_in,
                              void* d_out, int out_size, void* d_ws, size_t ws_size,
                              hipStream_t stream) {
    const float* n_feat = (const float*)d_in[0];   // [N,128]
    const float* e_w    = (const float*)d_in[1];   // [E,1]
    const float* W1     = (const float*)d_in[2];   // [256,256]
    const float* b1     = (const float*)d_in[3];
    const float* W2     = (const float*)d_in[4];   // [256,512]
    const float* b2     = (const float*)d_in[5];
    const float* W3     = (const float*)d_in[6];   // [64,512]
    const float* b3     = (const float*)d_in[7];
    const int*   src    = (const int*)d_in[8];
    const int*   dst    = (const int*)d_in[9];
    float* out = (float*)d_out;                    // [N,64]

    const int N = N_NODES, E = N_EDGES;

    // ---- workspace layout ----
    char* ws = (char*)d_ws;
    size_t off = 0;
    float* deg = (float*)(ws + off);              off = align256(off + (size_t)N * 4);
    __hip_bfloat16* W1b = (__hip_bfloat16*)(ws + off); off = align256(off + (size_t)256 * 256 * 2);
    __hip_bfloat16* W2b = (__hip_bfloat16*)(ws + off); off = align256(off + (size_t)256 * 512 * 2);
    __hip_bfloat16* W3b = (__hip_bfloat16*)(ws + off); off = align256(off + (size_t)64  * 512 * 2);
    float* agg = (float*)(ws + off);              off = align256(off + (size_t)N * 256 * 4);
    __hip_bfloat16* X = (__hip_bfloat16*)(ws + off);   off = align256(off + (size_t)N * 512 * 2);
    float* hA  = (float*)(ws + off);              off = align256(off + (size_t)N * 256 * 4);
    float* hB  = (float*)(ws + off);              off = align256(off + (size_t)N * 256 * 4);
    (void)ws_size; (void)n_in; (void)in_sizes; (void)out_size;

    const int TB = 256;

    // ---- degree + weight conversion ----
    hipMemsetAsync(deg, 0, (size_t)N * 4, stream);
    deg_kernel<<<(E + TB - 1) / TB, TB, 0, stream>>>(dst, deg, E);
    cvt_bf16_kernel<<<(256 * 256 + TB - 1) / TB, TB, 0, stream>>>(W1, W1b, 256 * 256);
    cvt_bf16_kernel<<<(256 * 512 + TB - 1) / TB, TB, 0, stream>>>(W2, W2b, 256 * 512);
    cvt_bf16_kernel<<<(64  * 512 + TB - 1) / TB, TB, 0, stream>>>(W3, W3b, 64 * 512);

    const int scatterBlocks = (E * 32 + TB - 1) / TB;   // one wave per edge

    // ---- layer 1: Din=128, K=256, Dout=256, ReLU ----
    hipMemsetAsync(agg, 0, (size_t)N * 128 * 4, stream);
    scatter_kernel<128><<<scatterBlocks, TB, 0, stream>>>(n_feat, e_w, src, dst, agg, E);
    build_x_kernel<128><<<((size_t)N * 256 + TB - 1) / TB, TB, 0, stream>>>(agg, n_feat, deg, X, N);
    {
        int waves = (N / 16) * (256 / 64);
        gemm_wmma_kernel<256, 256, 4, true>
            <<<(waves * 32 + TB - 1) / TB, TB, 0, stream>>>(X, W1b, b1, hA, N);
    }

    // ---- layer 2: Din=256, K=512, Dout=256, ReLU ----
    hipMemsetAsync(agg, 0, (size_t)N * 256 * 4, stream);
    scatter_kernel<256><<<scatterBlocks, TB, 0, stream>>>(hA, e_w, src, dst, agg, E);
    build_x_kernel<256><<<((size_t)N * 512 + TB - 1) / TB, TB, 0, stream>>>(agg, hA, deg, X, N);
    {
        int waves = (N / 16) * (256 / 64);
        gemm_wmma_kernel<512, 256, 4, true>
            <<<(waves * 32 + TB - 1) / TB, TB, 0, stream>>>(X, W2b, b2, hB, N);
    }

    // ---- layer 3: Din=256, K=512, Dout=64, no ReLU, -> d_out ----
    hipMemsetAsync(agg, 0, (size_t)N * 256 * 4, stream);
    scatter_kernel<256><<<scatterBlocks, TB, 0, stream>>>(hB, e_w, src, dst, agg, E);
    build_x_kernel<256><<<((size_t)N * 512 + TB - 1) / TB, TB, 0, stream>>>(agg, hB, deg, X, N);
    {
        int waves = (N / 16) * (64 / 64);
        gemm_wmma_kernel<512, 64, 4, false>
            <<<(waves * 32 + TB - 1) / TB, TB, 0, stream>>>(X, W3b, b3, out, N);
    }
}
